// RGCNN_Seg_66305705115960
// MI455X (gfx1250) — matrix operand order repaired and checked by reading, not compiled, not run on hardware
//
#include <hip/hip_runtime.h>

// RGCNN segmentation pipeline for MI455X (gfx1250), bf16 WMMA path.
// Gaussian-affinity Laplacian (bf16) + 3 Chebyshev conv layers.
// ~335 GFLOP, operands L2-resident => compute-bound => v_wmma_f32_16x16x32_bf16.
// GEMMs: 256x64 block tile (8 waves, 32x64 per wave), K-step 64, double-buffered
// dynamic LDS (92 KB of the WGP's 320 KB), 16 WMMAs per stage per wave,
// 1.5 ds_load_b128 per WMMA, one barrier per stage, compile-time FIN/F/KC.

#define NBATCH 16
#define MPTS   2048
#define NMROWS (NBATCH * MPTS)  // 32768
#define FIN0   6

typedef __attribute__((ext_vector_type(16))) __bf16 v16bf;
typedef __attribute__((ext_vector_type(8)))  __bf16 v8bf;
typedef __attribute__((ext_vector_type(8)))  float  v8f;

// LDS panels: As[2][256][72], Bt[2][64][72] (rows padded to 72 elems = 144 B
// so the 16-lane ds_load_b128 fragment reads spread across banks).
#define SMEM_BYTES ((2 * 256 * 72 + 2 * 64 * 72) * 2)  // 92160

static __device__ __forceinline__ v16bf cat16(v8bf lo, v8bf hi) {
  return __builtin_shufflevector(lo, hi, 0,1,2,3,4,5,6,7,8,9,10,11,12,13,14,15);
}

// ---------------------------------------------------------------- graph build

__global__ __launch_bounds__(256) void sq_kernel(const float* __restrict__ x,
                                                 float* __restrict__ sq) {
  int i = blockIdx.x * 256 + threadIdx.x;
  if (i >= NMROWS) return;
  const float* p = x + (size_t)i * FIN0;
  float s = 0.f;
#pragma unroll
  for (int f = 0; f < FIN0; ++f) s += p[f] * p[f];
  sq[i] = s;
}

__global__ __launch_bounds__(256) void degree_kernel(const float* __restrict__ x,
                                                     const float* __restrict__ sq,
                                                     float* __restrict__ dv) {
  int row = blockIdx.x;
  int n = row / MPTS;
  const float* xi = x + (size_t)row * FIN0;
  float xr[FIN0];
#pragma unroll
  for (int f = 0; f < FIN0; ++f) xr[f] = xi[f];
  float si = sq[row];
  const float* xb = x + (size_t)n * MPTS * FIN0;
  const float* sb = sq + (size_t)n * MPTS;
  float sum = 0.f;
  for (int j = threadIdx.x; j < MPTS; j += 256) {
    const float* xj = xb + (size_t)j * FIN0;
    float dot = 0.f;
#pragma unroll
    for (int f = 0; f < FIN0; ++f) dot += xr[f] * xj[f];
    sum += __expf(2.f * dot - si - sb[j]);
  }
  __shared__ float red[256];
  red[threadIdx.x] = sum;
  __syncthreads();
  for (int s = 128; s > 0; s >>= 1) {
    if (threadIdx.x < s) red[threadIdx.x] += red[threadIdx.x + s];
    __syncthreads();
  }
  if (threadIdx.x == 0) dv[row] = rsqrtf(red[0]);
}

__global__ __launch_bounds__(256) void laplacian_kernel(const float* __restrict__ x,
                                                        const float* __restrict__ sq,
                                                        const float* __restrict__ dv,
                                                        __bf16* __restrict__ Lb) {
  int row = blockIdx.x;
  int n = row / MPTS;
  int i = row % MPTS;
  const float* xi = x + (size_t)row * FIN0;
  float xr[FIN0];
#pragma unroll
  for (int f = 0; f < FIN0; ++f) xr[f] = xi[f];
  float si = sq[row];
  float di = dv[row];
  const float* xb = x + (size_t)n * MPTS * FIN0;
  const float* sb = sq + (size_t)n * MPTS;
  const float* db = dv + (size_t)n * MPTS;
  __bf16* Lrow = Lb + (size_t)n * MPTS * MPTS + (size_t)i * MPTS;
  for (int j = threadIdx.x; j < MPTS; j += 256) {
    const float* xj = xb + (size_t)j * FIN0;
    float dot = 0.f;
#pragma unroll
    for (int f = 0; f < FIN0; ++f) dot += xr[f] * xj[f];
    float a = __expf(2.f * dot - si - sb[j]);
    float v = -di * a * db[j];
    if (j == i) v += 1.f;
    Lrow[j] = (__bf16)v;
  }
}

__global__ __launch_bounds__(256) void f32_to_bf16_kernel(const float* __restrict__ s,
                                                          __bf16* __restrict__ d, size_t n) {
  size_t i = (size_t)blockIdx.x * 256 + threadIdx.x;
  if (i < n) d[i] = (__bf16)s[i];
}

// -------------------------------------------------------------- WMMA GEMMs
//
// Fragment layouts per CDNA5 ISA 7.12.2 (wave32, 16x16x32 bf16):
//  A (16x32): lane L -> row m=L%16, half h=L/16; elems = K[h*8..+7] ++ K[16+h*8..+7]
//  B (32x16): lane L -> col n=L%16, half h=L/16; elems = K[h*8..+7] ++ K[16+h*8..+7]
//  C/D f32:   vgpr r, lane L -> M=r+8*h, N=L%16

// Out[m,f] = alpha*(L @ X)[m,f] + beta*Z[m,f]   per batch (blockIdx.z)
template <int FIN>
__global__ __launch_bounds__(256) void cheb_mm_kernel(const __bf16* __restrict__ Lb,
                                                      const __bf16* __restrict__ X,
                                                      const __bf16* __restrict__ Z,
                                                      __bf16* __restrict__ Out,
                                                      float alpha, float beta) {
  extern __shared__ char smem[];
  auto As = reinterpret_cast<__bf16(*)[256][72]>(smem);
  auto Bt = reinterpret_cast<__bf16(*)[64][72]>(smem + 2 * 256 * 72 * 2);

  const int tid  = threadIdx.x;
  const int wave = tid >> 5, lane = tid & 31, half = lane >> 4, l16 = lane & 15;
  const int mBase = blockIdx.x * 256;
  const int nBase = blockIdx.y * 64;
  const size_t bL = (size_t)blockIdx.z * MPTS * MPTS;
  const size_t bX = (size_t)blockIdx.z * MPTS * FIN;

  const int bkk = tid >> 2, bc0 = (tid & 3) * 16;  // B fast path: 16 elems/thread

  uint4 aReg[8];   // A: one full 64-elem row per thread
  uint4 bRegV[2];
  __bf16 bRegE[16];

  auto loadStage = [&](int kt) {
    const int kBase = kt * 64;
    const __bf16* ap = Lb + bL + (size_t)(mBase + tid) * MPTS + kBase;
#pragma unroll
    for (int i = 0; i < 8; ++i) aReg[i] = reinterpret_cast<const uint4*>(ap)[i];
    if constexpr (FIN >= 64) {
      const __bf16* bp = X + bX + (size_t)(kBase + bkk) * FIN + nBase + bc0;
      bRegV[0] = reinterpret_cast<const uint4*>(bp)[0];
      bRegV[1] = reinterpret_cast<const uint4*>(bp)[1];
    } else {
#pragma unroll
      for (int i = 0; i < 16; ++i) {
        int idx = tid + i * 256;
        int c = idx & 63, kk = idx >> 6;
        bRegE[i] = (nBase + c < FIN) ? X[bX + (size_t)(kBase + kk) * FIN + (nBase + c)]
                                     : (__bf16)0.f;
      }
    }
  };
  auto storeStage = [&](int buf) {
    uint4* ad = reinterpret_cast<uint4*>(&As[buf][tid][0]);
#pragma unroll
    for (int i = 0; i < 8; ++i) ad[i] = aReg[i];
    if constexpr (FIN >= 64) {
      const __bf16* be = reinterpret_cast<const __bf16*>(bRegV);
#pragma unroll
      for (int e = 0; e < 16; ++e) Bt[buf][bc0 + e][bkk] = be[e];
    } else {
#pragma unroll
      for (int i = 0; i < 16; ++i) {
        int idx = tid + i * 256;
        int c = idx & 63, kk = idx >> 6;
        Bt[buf][c][kk] = bRegE[i];
      }
    }
  };

  v8f acc[2][4];
#pragma unroll
  for (int t = 0; t < 2; ++t)
#pragma unroll
    for (int j = 0; j < 4; ++j) acc[t][j] = (v8f){0.f,0.f,0.f,0.f,0.f,0.f,0.f,0.f};

  auto compute = [&](int buf) {
#pragma unroll
    for (int s = 0; s < 2; ++s) {
      v16bf af[2];
#pragma unroll
      for (int t = 0; t < 2; ++t) {
        const __bf16* ar = &As[buf][wave * 32 + t * 16 + l16][s * 32];
        af[t] = cat16(*reinterpret_cast<const v8bf*>(ar + half * 8),
                      *reinterpret_cast<const v8bf*>(ar + 16 + half * 8));
      }
#pragma unroll
      for (int j = 0; j < 4; ++j) {
        const __bf16* br = &Bt[buf][j * 16 + l16][s * 32];
        v16bf bf = cat16(*reinterpret_cast<const v8bf*>(br + half * 8),
                         *reinterpret_cast<const v8bf*>(br + 16 + half * 8));
#pragma unroll
        for (int t = 0; t < 2; ++t)
          acc[t][j] = __builtin_amdgcn_wmma_f32_16x16x32_bf16(false, af[t], false, bf,
                                                              (short)0, acc[t][j], false, false);
      }
    }
  };

  constexpr int NSTAGE = MPTS / 64;  // 32
  loadStage(0);
  storeStage(0);
  for (int kt = 0; kt < NSTAGE; ++kt) {
    __syncthreads();
    if (kt + 1 < NSTAGE) loadStage(kt + 1);
    compute(kt & 1);
    if (kt + 1 < NSTAGE) storeStage((kt + 1) & 1);
  }

#pragma unroll
  for (int t = 0; t < 2; ++t)
#pragma unroll
    for (int j = 0; j < 4; ++j) {
      int col = nBase + j * 16 + l16;
      if constexpr (FIN < 64) { if (col >= FIN) continue; }
#pragma unroll
      for (int r = 0; r < 8; ++r) {
        int m = mBase + wave * 32 + t * 16 + r + 8 * half;
        float v = alpha * acc[t][j][r];
        if (Z) v += beta * (float)Z[bX + (size_t)m * FIN + col];
        Out[bX + (size_t)m * FIN + col] = (__bf16)v;
      }
    }
}

// Y[NM,F] (+)= A[NM,FIN] @ Wk, Wk[f,c] = W[(f*KC + kc)*F + c] (fp32->bf16).
// MODE: 0 accumulate; 1 init (acc + bias); 2 final (ReLU -> bf16 Hout);
//       3 final (ReLU in-place on fp32 Y, used when Y = d_out).
template <int FIN, int F, int KC, int MODE>
__global__ __launch_bounds__(256) void dense_acc_kernel(const __bf16* __restrict__ A,
                                                        const float* __restrict__ W,
                                                        const float* __restrict__ bias,
                                                        float* __restrict__ Y,
                                                        __bf16* __restrict__ Hout,
                                                        int kc) {
  extern __shared__ char smem[];
  auto As = reinterpret_cast<__bf16(*)[256][72]>(smem);
  auto Bt = reinterpret_cast<__bf16(*)[64][72]>(smem + 2 * 256 * 72 * 2);

  const int tid  = threadIdx.x;
  const int wave = tid >> 5, lane = tid & 31, half = lane >> 4, l16 = lane & 15;
  const int mBase = blockIdx.x * 256;
  const int nBase = blockIdx.y * 64;

  const int bkk = tid >> 2, bc0 = (tid & 3) * 16;

  uint4  aReg[8];
  __bf16 aRegE[64];
  float4 wReg[4];

  auto loadStage = [&](int kt) {
    const int kBase = kt * 64;
    if constexpr (FIN % 64 == 0) {
      const __bf16* ap = A + (size_t)(mBase + tid) * FIN + kBase;
#pragma unroll
      for (int i = 0; i < 8; ++i) aReg[i] = reinterpret_cast<const uint4*>(ap)[i];
    } else {
#pragma unroll
      for (int i = 0; i < 64; ++i) {  // 16384 elems, guarded on K
        int idx = tid + i * 256;
        int r = idx >> 6, c = idx & 63;
        aRegE[i] = (kBase + c < FIN) ? A[(size_t)(mBase + r) * FIN + (kBase + c)]
                                     : (__bf16)0.f;
      }
    }
    int f = kBase + bkk;
    const float* wp = W + ((size_t)f * KC + kc) * F + nBase + bc0;
    if constexpr (FIN % 64 == 0) {
#pragma unroll
      for (int i = 0; i < 4; ++i) wReg[i] = reinterpret_cast<const float4*>(wp)[i];
    } else {
#pragma unroll
      for (int i = 0; i < 4; ++i)
        wReg[i] = (f < FIN) ? reinterpret_cast<const float4*>(wp)[i]
                            : (float4){0.f, 0.f, 0.f, 0.f};
    }
  };
  auto storeStage = [&](int buf) {
    if constexpr (FIN % 64 == 0) {
      uint4* ad = reinterpret_cast<uint4*>(&As[buf][tid][0]);
#pragma unroll
      for (int i = 0; i < 8; ++i) ad[i] = aReg[i];
    } else {
#pragma unroll
      for (int i = 0; i < 64; ++i) {
        int idx = tid + i * 256;
        int r = idx >> 6, c = idx & 63;
        As[buf][r][c] = aRegE[i];
      }
    }
    const float* we = reinterpret_cast<const float*>(wReg);
#pragma unroll
    for (int e = 0; e < 16; ++e) Bt[buf][bc0 + e][bkk] = (__bf16)we[e];
  };

  v8f acc[2][4];
#pragma unroll
  for (int t = 0; t < 2; ++t)
#pragma unroll
    for (int j = 0; j < 4; ++j) acc[t][j] = (v8f){0.f,0.f,0.f,0.f,0.f,0.f,0.f,0.f};

  auto compute = [&](int buf) {
#pragma unroll
    for (int s = 0; s < 2; ++s) {
      v16bf af[2];
#pragma unroll
      for (int t = 0; t < 2; ++t) {
        const __bf16* ar = &As[buf][wave * 32 + t * 16 + l16][s * 32];
        af[t] = cat16(*reinterpret_cast<const v8bf*>(ar + half * 8),
                      *reinterpret_cast<const v8bf*>(ar + 16 + half * 8));
      }
#pragma unroll
      for (int j = 0; j < 4; ++j) {
        const __bf16* br = &Bt[buf][j * 16 + l16][s * 32];
        v16bf bf = cat16(*reinterpret_cast<const v8bf*>(br + half * 8),
                         *reinterpret_cast<const v8bf*>(br + 16 + half * 8));
#pragma unroll
        for (int t = 0; t < 2; ++t)
          acc[t][j] = __builtin_amdgcn_wmma_f32_16x16x32_bf16(false, af[t], false, bf,
                                                              (short)0, acc[t][j], false, false);
      }
    }
  };

  constexpr int NSTAGE = (FIN + 63) / 64;
  loadStage(0);
  storeStage(0);
  for (int kt = 0; kt < NSTAGE; ++kt) {
    __syncthreads();
    if (kt + 1 < NSTAGE) loadStage(kt + 1);
    compute(kt & 1);
    if (kt + 1 < NSTAGE) storeStage((kt + 1) & 1);
  }

#pragma unroll
  for (int t = 0; t < 2; ++t)
#pragma unroll
    for (int j = 0; j < 4; ++j) {
      int col = nBase + j * 16 + l16;  // F multiple of 64: always in range
#pragma unroll
      for (int r = 0; r < 8; ++r) {
        int m = mBase + wave * 32 + t * 16 + r + 8 * half;
        size_t idx = (size_t)m * F + col;
        if constexpr (MODE == 0) {
          Y[idx] += acc[t][j][r];
        } else if constexpr (MODE == 1) {
          Y[idx] = acc[t][j][r] + bias[col];
        } else if constexpr (MODE == 2) {
          Hout[idx] = (__bf16)fmaxf(Y[idx] + acc[t][j][r], 0.f);
        } else {
          Y[idx] = fmaxf(Y[idx] + acc[t][j][r], 0.f);
        }
      }
    }
}

// ----------------------------------------------------------------- host side

template <int FIN, int F, int KC, bool LAST>
static void run_layer(const __bf16* Lb, __bf16* T0, __bf16* tb0, __bf16* tb1, __bf16* tb2,
                      const float* W, const float* bias, float* Y, __bf16* Hout,
                      hipStream_t stream) {
  dim3 gd(NMROWS / 256, F / 64, 1);
  dim3 gc(MPTS / 256, (FIN + 63) / 64, NBATCH);

  dense_acc_kernel<FIN, F, KC, 1><<<gd, 256, SMEM_BYTES, stream>>>(T0, W, bias, Y, nullptr, 0);

  __bf16* bufs[3] = {tb0, tb1, tb2};
  __bf16* tp1 = T0;       // T_{k-1}
  __bf16* tp2 = nullptr;  // T_{k-2}
  for (int k = 1; k < KC; ++k) {
    __bf16* tk = bufs[(k - 1) % 3];
    if (k == 1)
      cheb_mm_kernel<FIN><<<gc, 256, SMEM_BYTES, stream>>>(Lb, tp1, (const __bf16*)nullptr,
                                                           tk, 1.f, 0.f);
    else
      cheb_mm_kernel<FIN><<<gc, 256, SMEM_BYTES, stream>>>(Lb, tp1, tp2, tk, 2.f, -1.f);
    if (k == KC - 1) {
      if (LAST)
        dense_acc_kernel<FIN, F, KC, 3><<<gd, 256, SMEM_BYTES, stream>>>(tk, W, nullptr, Y,
                                                                         nullptr, k);
      else
        dense_acc_kernel<FIN, F, KC, 2><<<gd, 256, SMEM_BYTES, stream>>>(tk, W, nullptr, Y,
                                                                         Hout, k);
    } else {
      dense_acc_kernel<FIN, F, KC, 0><<<gd, 256, SMEM_BYTES, stream>>>(tk, W, nullptr, Y,
                                                                       nullptr, k);
    }
    tp2 = tp1;
    tp1 = tk;
  }
}

extern "C" void kernel_launch(void* const* d_in, const int* in_sizes, int n_in,
                              void* d_out, int out_size, void* d_ws, size_t ws_size,
                              hipStream_t stream) {
  (void)in_sizes; (void)n_in; (void)out_size; (void)ws_size;
  const float* x  = (const float*)d_in[0];
  // d_in[1] = cat (unused by the reference computation)
  const float* W0 = (const float*)d_in[2];
  const float* B0 = (const float*)d_in[3];
  const float* W1 = (const float*)d_in[4];
  const float* B1 = (const float*)d_in[5];
  const float* W2 = (const float*)d_in[6];
  const float* B2 = (const float*)d_in[7];
  float* out = (float*)d_out;  // [16, 2048, 1024] fp32

  char* ws = (char*)d_ws;
  size_t off = 0;
  auto alloc = [&](size_t bytes) -> void* {
    void* p = ws + off;
    off += (bytes + 255) & ~(size_t)255;
    return p;
  };
  __bf16* Lb   = (__bf16*)alloc((size_t)NBATCH * MPTS * MPTS * 2);  // 134 MB
  float*  sq   = (float*)alloc((size_t)NMROWS * 4);
  float*  dv   = (float*)alloc((size_t)NMROWS * 4);
  __bf16* Hbuf = (__bf16*)alloc((size_t)NMROWS * 512 * 2);
  __bf16* Tb0  = (__bf16*)alloc((size_t)NMROWS * 512 * 2);
  __bf16* Tb1  = (__bf16*)alloc((size_t)NMROWS * 512 * 2);
  __bf16* Tb2  = (__bf16*)alloc((size_t)NMROWS * 512 * 2);
  float*  ybuf = (float*)alloc((size_t)NMROWS * 512 * 4);           // 67 MB

  sq_kernel<<<NMROWS / 256, 256, 0, stream>>>(x, sq);
  degree_kernel<<<NMROWS, 256, 0, stream>>>(x, sq, dv);
  laplacian_kernel<<<NMROWS, 256, 0, stream>>>(x, sq, dv, Lb);
  f32_to_bf16_kernel<<<(int)(((size_t)NMROWS * FIN0 + 255) / 256), 256, 0, stream>>>(
      x, Hbuf, (size_t)NMROWS * FIN0);

  run_layer<6,   128,  6, false>(Lb, Hbuf, Tb0, Tb1, Tb2, W0, B0, ybuf, Hbuf, stream);
  run_layer<128, 512,  5, false>(Lb, Hbuf, Tb0, Tb1, Tb2, W1, B1, ybuf, Hbuf, stream);
  run_layer<512, 1024, 3, true >(Lb, Hbuf, Tb0, Tb1, Tb2, W2, B2, out,  nullptr, stream);
}
